// Net_55774445305918
// MI455X (gfx1250) — matrix operand order repaired and checked by reading, not compiled, not run on hardware
//
#include <hip/hip_runtime.h>
#include <hip/hip_bf16.h>
#include <cstdint>

// ---------------------------------------------------------------------------
// CDNA5 / gfx1250 keypoint-encoder forward pass.
// Convs = implicit-GEMM on v_wmma_f32_16x16x32_f16; weight panels staged in
// LDS K-major so B fragments are single 32B ds_load_b128 pairs; conv2 weights
// DMA'd into LDS with the Tensor Data Mover (tensor_load_to_lds).
// ---------------------------------------------------------------------------

typedef __attribute__((ext_vector_type(16))) _Float16 v16h;
typedef __attribute__((ext_vector_type(8)))  float    v8f;
typedef __attribute__((ext_vector_type(4))) uint32_t  v4u;
typedef __attribute__((ext_vector_type(8))) int       v8i;
typedef __attribute__((ext_vector_type(4))) int       v4i;

#define BATCH 512
#define HW    80
#define C0    3
#define C1    20
#define H1    78
#define C2    30
#define H2    38
#define NK    3
#define H3    18
#define BN_EPS   1e-5f
#define LOSS_EPS 1e-7f

__device__ __forceinline__ v8f wmma_f16(v16h a, v16h b, v8f c) {
  return __builtin_amdgcn_wmma_f32_16x16x32_f16(false, a, false, b, (short)0, c,
                                                false, false);
}

__device__ __forceinline__ int iclampi(int v, int lo, int hi) {
  return v < lo ? lo : (v > hi ? hi : v);
}

__device__ float blockReduceSum(float v, float* s) {
  int t = threadIdx.x;
  __syncthreads();
  s[t] = v;
  __syncthreads();
  for (int st = 128; st > 0; st >>= 1) {
    if (t < st) s[t] += s[t + st];
    __syncthreads();
  }
  float r = s[0];
  __syncthreads();
  return r;
}

__device__ float blockReduceMax(float v, float* s) {
  int t = threadIdx.x;
  __syncthreads();
  s[t] = v;
  __syncthreads();
  for (int st = 128; st > 0; st >>= 1) {
    if (t < st) s[t] = fmaxf(s[t], s[t + st]);
    __syncthreads();
  }
  float r = s[0];
  __syncthreads();
  return r;
}

// ---------------------------------------------------------------------------
// conv1: [B,3,80,80] -(x-0.33, 3x3 s1, relu)-> x2 f16 [B,20,78,78]
// ---------------------------------------------------------------------------
__global__ __launch_bounds__(128) void conv1_kernel(
    const float* __restrict__ x, const float* __restrict__ w1,
    const float* __restrict__ bias1, _Float16* __restrict__ x2) {
  __shared__ __align__(32) _Float16 s_w[32 * 32];  // [n][k]
  for (int i = threadIdx.x; i < 32 * 32; i += 128) {
    int n = i >> 5, k = i & 31;
    float v = 0.f;
    if (k < 27 && n < C1) {
      int c = k / 9, r = k % 9;
      v = w1[((n * C0 + c) * 3 + r / 3) * 3 + (r % 3)];
    }
    s_w[i] = (_Float16)v;
  }
  __syncthreads();

  int lane = threadIdx.x & 31;
  int wave = threadIdx.x >> 5;
  int Mbase = (blockIdx.x * 4 + wave) * 16;

  int mrow = Mbase + (lane & 15);
  int b  = mrow / (H1 * H1);
  int rm = mrow % (H1 * H1);
  int oh = rm / H1, ow = rm % H1;
  const float* xb = x + (long)b * C0 * (HW * HW) + oh * HW + ow;
  __builtin_prefetch(xb, 0, 1);

  // koff is a literal at each call site -> k, c, kh, kw constant-fold and
  // every load becomes base + immediate offset.
  auto loadA = [&](int koff) {
    v16h a;
#pragma unroll
    for (int e = 0; e < 16; ++e) {
      int k = e + koff;
      float v = 0.f;
      if (k < 27) {
        int c = k / 9, r9 = k % 9;
        v = xb[c * (HW * HW) + (r9 / 3) * HW + (r9 % 3)] - 0.33f;
      }
      a[e] = (_Float16)v;
    }
    return a;
  };
  v16h a = (lane < 16) ? loadA(0) : loadA(16);

  int bn = lane & 15;
  int bko = (lane >> 4) << 4;
  v16h b0 = *reinterpret_cast<const v16h*>(&s_w[bn * 32 + bko]);
  v16h b1 = *reinterpret_cast<const v16h*>(&s_w[(16 + bn) * 32 + bko]);

  v8f c0 = {}, c1 = {};
  c0 = wmma_f16(a, b0, c0);
  c1 = wmma_f16(a, b1, c1);

  float bias_lo = bias1[bn];
  int n1 = 16 + bn;
  float bias_hi = (n1 < C1) ? bias1[n1] : 0.f;

  int m0 = Mbase + (lane >> 4) * 8;
  int bb = m0 / (H1 * H1);
  int rr = m0 % (H1 * H1);
  _Float16* out_lo = x2 + ((long)bb * C1 + bn) * (H1 * H1);
  _Float16* out_hi = x2 + ((long)bb * C1 + n1) * (H1 * H1);
#pragma unroll
  for (int r = 0; r < 8; ++r) {
    if (rr == H1 * H1) {
      bb++;
      rr = 0;
      out_lo = x2 + ((long)bb * C1 + bn) * (H1 * H1);
      out_hi = x2 + ((long)bb * C1 + n1) * (H1 * H1);
    }
    out_lo[rr] = (_Float16)fmaxf(c0[r] + bias_lo, 0.f);
    if (n1 < C1) out_hi[rr] = (_Float16)fmaxf(c1[r] + bias_hi, 0.f);
    rr++;
  }
}

// ---------------------------------------------------------------------------
// conv2: x2 f16 [B,20,78,78] -(3x3 s2)-> h f32 [B,30,38,38]  (pre-BN)
// Raw w2 (5400 f32) staged into LDS with the Tensor Data Mover.
// ---------------------------------------------------------------------------
__global__ __launch_bounds__(128) void conv2_kernel(
    const _Float16* __restrict__ x2, const float* __restrict__ w2,
    const float* __restrict__ bias2, float* __restrict__ h) {
  __shared__ float s_stage[C2 * C1 * 9];            // 5400 f32, TDM target
  __shared__ __align__(32) _Float16 s_w[32 * 192];  // [n][k] padded panel

#if defined(__HIP_DEVICE_COMPILE__)
  if (threadIdx.x == 0) {
    uint32_t lds_off = (uint32_t)(uintptr_t)(&s_stage[0]);
    uint64_t ga = (uint64_t)(uintptr_t)w2;
    v4u g0 = {0u, 0u, 0u, 0u};
    g0[0] = 1u;                                       // count=1, user mode
    g0[1] = lds_off;                                  // lds_addr
    g0[2] = (uint32_t)ga;                             // global_addr[31:0]
    g0[3] = ((uint32_t)(ga >> 32) & 0x01FFFFFFu) | (2u << 30);  // addr hi, type=2
    v8i g1 = {0, 0, 0, 0, 0, 0, 0, 0};
    g1[0] = (int)(2u << 16);                          // data_size = 4 bytes
    g1[1] = (int)((5400u & 0xFFFFu) << 16);           // tensor_dim0[15:0]
    g1[2] = (int)(1u << 16);                          // tensor_dim1 = 1
    g1[3] = (int)((5400u & 0xFFFFu) << 16);           // tile_dim0 = 5400
    g1[4] = 1;                                        // tile_dim1 = 1
    g1[5] = 5400;                                     // tensor_dim0_stride
    v4i gz = {0, 0, 0, 0};
#if __clang_major__ >= 23
    v8i gz8 = {0, 0, 0, 0, 0, 0, 0, 0};
    __builtin_amdgcn_tensor_load_to_lds(g0, g1, gz, gz, gz8, 0);
#else
    __builtin_amdgcn_tensor_load_to_lds(g0, g1, gz, gz, 0);
#endif
    __builtin_amdgcn_s_wait_tensorcnt(0);
  }
#else
  for (int i = threadIdx.x; i < C2 * C1 * 9; i += 128) s_stage[i] = w2[i];
#endif
  __syncthreads();

  for (int i = threadIdx.x; i < 32 * 192; i += 128) {
    int n = i / 192, k = i % 192;
    float v = 0.f;
    if (k < 180 && n < C2) {
      int c = k / 9, r = k % 9;
      v = s_stage[((n * C1 + c) * 3 + r / 3) * 3 + (r % 3)];
    }
    s_w[i] = (_Float16)v;
  }
  __syncthreads();

  int lane = threadIdx.x & 31;
  int wave = threadIdx.x >> 5;
  int Mbase = (blockIdx.x * 4 + wave) * 16;

  int mrow = Mbase + (lane & 15);
  int b  = mrow / (H2 * H2);
  int rm = mrow % (H2 * H2);
  int oh = rm / H2, ow = rm % H2;
  const _Float16* xb = x2 + (long)b * C1 * (H1 * H1) + (oh * 2) * H1 + (ow * 2);
  __builtin_prefetch(xb, 0, 1);

  int bn = lane & 15;
  int bko = (lane >> 4) << 4;

  auto loadA = [&](int kbase, int koff) {
    v16h a;
#pragma unroll
    for (int e = 0; e < 16; ++e) {
      int k = kbase + e + koff;
      _Float16 v = (_Float16)0.f;
      if (k < 180) {
        int c = k / 9, r9 = k % 9;
        v = xb[c * (H1 * H1) + (r9 / 3) * H1 + (r9 % 3)];
      }
      a[e] = v;
    }
    return a;
  };

  v8f c0 = {}, c1 = {};
#pragma unroll
  for (int kc = 0; kc < 6; ++kc) {
    int kbase = kc * 32;
    v16h a = (lane < 16) ? loadA(kbase, 0) : loadA(kbase, 16);
    v16h b0 = *reinterpret_cast<const v16h*>(&s_w[bn * 192 + kbase + bko]);
    v16h b1 = *reinterpret_cast<const v16h*>(&s_w[(16 + bn) * 192 + kbase + bko]);
    c0 = wmma_f16(a, b0, c0);
    c1 = wmma_f16(a, b1, c1);
  }

  float bias_lo = bias2[bn];
  int n1 = 16 + bn;
  float bias_hi = (n1 < C2) ? bias2[n1] : 0.f;

  int m0 = Mbase + (lane >> 4) * 8;
  int bb = m0 / (H2 * H2);
  int rr = m0 % (H2 * H2);
  float* out_lo = h + ((long)bb * C2 + bn) * (H2 * H2);
  float* out_hi = h + ((long)bb * C2 + n1) * (H2 * H2);
#pragma unroll
  for (int r = 0; r < 8; ++r) {
    if (rr == H2 * H2) {
      bb++;
      rr = 0;
      out_lo = h + ((long)bb * C2 + bn) * (H2 * H2);
      out_hi = h + ((long)bb * C2 + n1) * (H2 * H2);
    }
    out_lo[rr] = c0[r] + bias_lo;
    if (n1 < C2) out_hi[rr] = c1[r] + bias_hi;
    rr++;
  }
}

// ---------------------------------------------------------------------------
// BatchNorm stats: deterministic two-stage reduction.
// ---------------------------------------------------------------------------
__global__ __launch_bounds__(256) void bn_partial_kernel(
    const float* __restrict__ h, float* __restrict__ psum,
    float* __restrict__ psq) {
  __shared__ float s1[256], s2[256];
  int c = blockIdx.x;
  int chunk = blockIdx.y;
  const int per = (BATCH * H2 * H2) / 256;  // 2888
  float a = 0.f, b = 0.f;
  for (int j = chunk * per + threadIdx.x; j < (chunk + 1) * per; j += 256) {
    int bb = j / (H2 * H2), s = j % (H2 * H2);
    float v = h[((long)bb * C2 + c) * (H2 * H2) + s];
    a += v;
    b += v * v;
  }
  s1[threadIdx.x] = a;
  s2[threadIdx.x] = b;
  __syncthreads();
  for (int st = 128; st > 0; st >>= 1) {
    if (threadIdx.x < st) {
      s1[threadIdx.x] += s1[threadIdx.x + st];
      s2[threadIdx.x] += s2[threadIdx.x + st];
    }
    __syncthreads();
  }
  if (threadIdx.x == 0) {
    psum[c * 256 + chunk] = s1[0];
    psq[c * 256 + chunk] = s2[0];
  }
}

__global__ __launch_bounds__(256) void bn_final_kernel(
    const float* __restrict__ psum, const float* __restrict__ psq,
    float* __restrict__ stats) {
  __shared__ float s_red[256];
  int c = blockIdx.x;
  float a = blockReduceSum(psum[c * 256 + threadIdx.x], s_red);
  float b = blockReduceSum(psq[c * 256 + threadIdx.x], s_red);
  if (threadIdx.x == 0) {
    float Nc = (float)(BATCH * H2 * H2);
    float mu = a / Nc;
    float var = b / Nc - mu * mu;
    stats[c] = mu;
    stats[C2 + c] = rsqrtf(var + BN_EPS);
  }
}

// ---------------------------------------------------------------------------
// conv3: h (BN+relu fused into A-gather) -(3x3 s2)-> dense f32 [B,3,18,18]
// ---------------------------------------------------------------------------
__global__ __launch_bounds__(128) void conv3_kernel(
    const float* __restrict__ h, const float* __restrict__ stats,
    const float* __restrict__ gamma, const float* __restrict__ beta,
    const float* __restrict__ w3, const float* __restrict__ bias3,
    float* __restrict__ dense) {
  __shared__ __align__(32) _Float16 s_w[16 * 288];  // [n][k]
  __shared__ float s_mu[C2], s_inv[C2], s_g[C2], s_b[C2];
  for (int i = threadIdx.x; i < C2; i += 128) {
    s_mu[i] = stats[i];
    s_inv[i] = stats[C2 + i];
    s_g[i] = gamma[i];
    s_b[i] = beta[i];
  }
  for (int i = threadIdx.x; i < 16 * 288; i += 128) {
    int n = i / 288, k = i % 288;
    float v = 0.f;
    if (k < 270 && n < NK) {
      int c = k / 9, r = k % 9;
      v = w3[((n * C2 + c) * 3 + r / 3) * 3 + (r % 3)];
    }
    s_w[i] = (_Float16)v;
  }
  __syncthreads();

  int lane = threadIdx.x & 31;
  int wave = threadIdx.x >> 5;
  int Mbase = (blockIdx.x * 4 + wave) * 16;

  int mrow = Mbase + (lane & 15);
  int b  = mrow / (H3 * H3);
  int rm = mrow % (H3 * H3);
  int oh = rm / H3, ow = rm % H3;
  const float* hb = h + (long)b * C2 * (H2 * H2) + (oh * 2) * H2 + (ow * 2);
  __builtin_prefetch(hb, 0, 1);

  int bn = lane & 15;
  int bko = (lane >> 4) << 4;

  auto loadA = [&](int kbase, int koff) {
    v16h a;
#pragma unroll
    for (int e = 0; e < 16; ++e) {
      int k = kbase + e + koff;
      float v = 0.f;
      if (k < 270) {
        int c = k / 9, r9 = k % 9;
        float hv = hb[c * (H2 * H2) + (r9 / 3) * H2 + (r9 % 3)];
        v = fmaxf((hv - s_mu[c]) * s_inv[c] * s_g[c] + s_b[c], 0.f);
      }
      a[e] = (_Float16)v;
    }
    return a;
  };

  v8f c0 = {};
#pragma unroll
  for (int kc = 0; kc < 9; ++kc) {
    int kbase = kc * 32;
    v16h a = (lane < 16) ? loadA(kbase, 0) : loadA(kbase, 16);
    v16h bf = *reinterpret_cast<const v16h*>(&s_w[bn * 288 + kbase + bko]);
    c0 = wmma_f16(a, bf, c0);
  }

  if (bn < NK) {
    float bias = bias3[bn];
    int m0 = Mbase + (lane >> 4) * 8;
    int bb = m0 / (H3 * H3);
    int rr = m0 % (H3 * H3);
    float* outp = dense + ((long)bb * NK + bn) * (H3 * H3);
#pragma unroll
    for (int r = 0; r < 8; ++r) {
      if (rr == H3 * H3) {
        bb++;
        rr = 0;
        outp = dense + ((long)bb * NK + bn) * (H3 * H3);
      }
      outp[rr] = c0[r] + bias;
      rr++;
    }
  }
}

// ---------------------------------------------------------------------------
// Per (b,k): bilinear 18->80, spatial softmax, soft-argmax.
// ---------------------------------------------------------------------------
__global__ __launch_bounds__(256) void softmax_kernel(
    const float* __restrict__ dense, float* __restrict__ kp,
    float* __restrict__ map_out) {
  __shared__ float s_d[H3 * H3];
  __shared__ float s_up[HW * HW];
  __shared__ float s_red[256];
  int bk = blockIdx.x;
  int tid = threadIdx.x;
  for (int i = tid; i < H3 * H3; i += 256)
    s_d[i] = dense[(long)bk * (H3 * H3) + i];
  __syncthreads();

  const float scale = (float)H3 / (float)HW;  // 0.225
  float mx = -3.4e38f;
  for (int p = tid; p < HW * HW; p += 256) {
    int oy = p / HW, ox = p % HW;
    float sy = ((float)oy + 0.5f) * scale - 0.5f;
    float sx = ((float)ox + 0.5f) * scale - 0.5f;
    int y0 = (int)floorf(sy);
    int x0 = (int)floorf(sx);
    float wy = sy - (float)y0;
    float wx = sx - (float)x0;
    int ya = iclampi(y0, 0, H3 - 1), yb = iclampi(y0 + 1, 0, H3 - 1);
    int xa = iclampi(x0, 0, H3 - 1), xb = iclampi(x0 + 1, 0, H3 - 1);
    float v = (1.f - wy) * ((1.f - wx) * s_d[ya * H3 + xa] + wx * s_d[ya * H3 + xb]) +
              wy * ((1.f - wx) * s_d[yb * H3 + xa] + wx * s_d[yb * H3 + xb]);
    s_up[p] = v;
    mx = fmaxf(mx, v);
  }
  mx = blockReduceMax(mx, s_red);

  float acc = 0.f, accx = 0.f, accy = 0.f;
  for (int p = tid; p < HW * HW; p += 256) {
    int oy = p / HW, ox = p % HW;
    float e = __expf(s_up[p] - mx);
    s_up[p] = e;
    acc += e;
    accx += e * (float)ox;
    accy += e * (float)oy;
  }
  float S = blockReduceSum(acc, s_red);
  float SX = blockReduceSum(accx, s_red);
  float SY = blockReduceSum(accy, s_red);
  float inv = 1.f / S;

  if (map_out) {
    for (int p = tid; p < HW * HW; p += 256)
      map_out[(long)bk * (HW * HW) + p] = s_up[p] * inv;
  }
  if (tid == 0) {
    kp[bk * 2 + 0] = SX * inv;  // col coordinate first, per reference
    kp[bk * 2 + 1] = SY * inv;
  }
}

// ---------------------------------------------------------------------------
__global__ __launch_bounds__(256) void imgchange_kernel(
    const float* __restrict__ f1, const float* __restrict__ f2,
    float* __restrict__ chg) {
  int idx = blockIdx.x * 256 + threadIdx.x;
  if (idx >= BATCH * HW * HW) return;
  int b = idx / (HW * HW), p = idx % (HW * HW);
  float s = 0.f;
#pragma unroll
  for (int c = 0; c < C0; ++c) {
    float d = fabsf(f1[((long)b * C0 + c) * (HW * HW) + p] -
                    f2[((long)b * C0 + c) * (HW * HW) + p]);
    s += (d > 0.f) ? 1.f : 0.f;
  }
  chg[idx] = s;
}

__global__ __launch_bounds__(256) void overlap_kernel(
    const float* __restrict__ map1, const float* __restrict__ chg,
    float* __restrict__ ovl) {
  __shared__ float s_red[256];
  int bk = blockIdx.x;
  int b = bk / NK;
  float acc = 0.f;
  for (int p = threadIdx.x; p < HW * HW; p += 256)
    acc += map1[(long)bk * (HW * HW) + p] * chg[(long)b * (HW * HW) + p];
  acc = blockReduceSum(acc, s_red);
  if (threadIdx.x == 0) ovl[bk] = acc;
}

__global__ __launch_bounds__(256) void imgsum_kernel(
    const float* __restrict__ chg, float* __restrict__ isum) {
  __shared__ float s_red[256];
  int b = blockIdx.x;
  float acc = 0.f;
  for (int p = threadIdx.x; p < HW * HW; p += 256)
    acc += chg[(long)b * (HW * HW) + p];
  acc = blockReduceSum(acc, s_red);
  if (threadIdx.x == 0) isum[b] = acc;
}

// ---------------------------------------------------------------------------
__global__ __launch_bounds__(256) void finalize_kernel(
    const float* __restrict__ kp1, const float* __restrict__ kp1p,
    const float* __restrict__ ovl, const float* __restrict__ isum,
    float* __restrict__ losses) {
  __shared__ float s_red[256];
  int tid = threadIdx.x;

  float acc = 0.f;
  for (int i = tid; i < BATCH * NK; i += 256) {
    float d = (kp1[i * 2] - kp1p[i * 2]) + (kp1[i * 2 + 1] - kp1p[i * 2 + 1]);
    acc += d * d;
  }
  float kc = blockReduceSum(acc, s_red) / (float)(BATCH * NK);

  float msum = 0.f;
  for (int b = tid; b < BATCH; b += 256)
    msum += (isum[b] / (float)(HW * HW) > 0.f) ? 1.f : 0.f;
  msum = blockReduceSum(msum, s_red);

  float ssum = 0.f;
  for (int i = tid; i < BATCH * NK; i += 256) {
    int b = i / NK;
    float mask = (isum[b] / (float)(HW * HW) > 0.f) ? 1.f : 0.f;
    ssum += mask * (-logf(LOSS_EPS + ovl[i]));
  }
  ssum = blockReduceSum(ssum, s_red);
  float sc = ssum / fmaxf(msum * (float)NK, 1.f);

  float kvacc = 0.f;
  for (int t = tid; t < BATCH * NK * 2; t += 256) {
    int b = t / (NK * 2);
    int r = t % (NK * 2);
    int i = r / 2, j = r % 2;
    if (i != j) {
      float dx = kp1[(b * NK + i) * 2 + 0] - kp1[(b * NK + j) * 2 + 0];
      float dy = kp1[(b * NK + i) * 2 + 1] - kp1[(b * NK + j) * 2 + 1];
      kvacc += fmaxf(dx * dx + dy * dy, 0.f);
    }
  }
  kvacc = blockReduceSum(kvacc, s_red);
  float kv = kvacc / (float)(NK * NK * BATCH);

  if (tid == 0) {
    losses[0] = kv;
    losses[1] = kc;
    losses[2] = sc;
  }
}

// ---------------------------------------------------------------------------
extern "C" void kernel_launch(void* const* d_in, const int* in_sizes, int n_in,
                              void* d_out, int out_size, void* d_ws,
                              size_t ws_size, hipStream_t stream) {
  (void)in_sizes; (void)n_in; (void)out_size; (void)ws_size;
  const float* first  = (const float*)d_in[0];
  const float* firstp = (const float*)d_in[1];
  const float* second = (const float*)d_in[2];
  const float* w1 = (const float*)d_in[3];
  const float* b1 = (const float*)d_in[4];
  const float* w2 = (const float*)d_in[5];
  const float* b2 = (const float*)d_in[6];
  const float* gamma = (const float*)d_in[7];
  const float* beta = (const float*)d_in[8];
  const float* w3 = (const float*)d_in[9];
  const float* b3 = (const float*)d_in[10];
  float* out = (float*)d_out;

  char* ws = (char*)d_ws;
  size_t off = 0;
  auto carve = [&](size_t bytes) -> void* {
    off = (off + 255) & ~(size_t)255;
    void* p = ws + off;
    off += bytes;
    return p;
  };
  _Float16* x2 = (_Float16*)carve((size_t)BATCH * C1 * H1 * H1 * sizeof(_Float16));
  float* h     = (float*)carve((size_t)BATCH * C2 * H2 * H2 * sizeof(float));
  float* dense = (float*)carve((size_t)BATCH * NK * H3 * H3 * sizeof(float));
  float* psum  = (float*)carve((size_t)C2 * 256 * sizeof(float));
  float* psq   = (float*)carve((size_t)C2 * 256 * sizeof(float));
  float* stats = (float*)carve((size_t)2 * C2 * sizeof(float));
  float* ovl   = (float*)carve((size_t)BATCH * NK * sizeof(float));
  float* isum  = (float*)carve((size_t)BATCH * sizeof(float));

  float* kp1  = out;
  float* kp2  = out + (size_t)BATCH * NK * 2;
  float* kp1p = out + (size_t)2 * BATCH * NK * 2;
  float* map1 = out + (size_t)3 * BATCH * NK * 2;
  float* map2 = map1 + (size_t)BATCH * NK * HW * HW;
  float* chg  = map2 + (size_t)BATCH * NK * HW * HW;
  float* losses = chg + (size_t)BATCH * HW * HW;

  auto encode = [&](const float* xin, float* kp, float* map) {
    conv1_kernel<<<(BATCH * H1 * H1 / 16) / 4, 128, 0, stream>>>(xin, w1, b1, x2);
    conv2_kernel<<<(BATCH * H2 * H2 / 16) / 4, 128, 0, stream>>>(x2, w2, b2, h);
    bn_partial_kernel<<<dim3(C2, 256), 256, 0, stream>>>(h, psum, psq);
    bn_final_kernel<<<C2, 256, 0, stream>>>(psum, psq, stats);
    conv3_kernel<<<(BATCH * H3 * H3 / 16) / 4, 128, 0, stream>>>(
        h, stats, gamma, beta, w3, b3, dense);
    softmax_kernel<<<BATCH * NK, 256, 0, stream>>>(dense, kp, map);
  };

  encode(first, kp1, map1);
  encode(second, kp2, map2);
  encode(firstp, kp1p, nullptr);

  imgchange_kernel<<<(BATCH * HW * HW + 255) / 256, 256, 0, stream>>>(first,
                                                                      second, chg);
  overlap_kernel<<<BATCH * NK, 256, 0, stream>>>(map1, chg, ovl);
  imgsum_kernel<<<BATCH, 256, 0, stream>>>(chg, isum);
  finalize_kernel<<<1, 256, 0, stream>>>(kp1, kp1p, ovl, isum, losses);
}
